// Decoder_12103217840719
// MI455X (gfx1250) — matrix operand order repaired and checked by reading, not compiled, not run on hardware
//
#include <hip/hip_runtime.h>
#include <hip/hip_bf16.h>
#include <cstdint>
#include <cstddef>

// Problem dims
#define V_  32000
#define E_  512
#define U_  1024
#define B_  16
#define S_  128
#define G4  (4*U_)   // 4096
#define M_  (B_*S_)  // 2048

typedef __attribute__((ext_vector_type(16))) __bf16 v16bf;
typedef __attribute__((ext_vector_type(8)))  float  v8f;

// ---- CDNA5 async global->LDS copy (ASYNCcnt), with safe fallback ----
#if defined(__has_builtin)
#  if __has_builtin(__builtin_amdgcn_global_load_async_to_lds_b128)
#    define HAVE_ASYNC_LDS 1
#  endif
#endif
#ifndef HAVE_ASYNC_LDS
#  define HAVE_ASYNC_LDS 0
#endif

#if HAVE_ASYNC_LDS
// Builtin expects pointers to gcc-style int vectors (b128 payload):
//   (v4i AS1* gsrc, v4i AS3* ldst, imm int offset, imm int cpol)
typedef int v4i_ __attribute__((vector_size(16)));
typedef __attribute__((address_space(1))) v4i_* as1_v4i_p;
typedef __attribute__((address_space(3))) v4i_* as3_v4i_p;
#endif

__device__ __forceinline__ void copy16_to_lds(const void* g, void* l) {
#if HAVE_ASYNC_LDS
  __builtin_amdgcn_global_load_async_to_lds_b128(
      (as1_v4i_p)(uintptr_t)g, (as3_v4i_p)(uintptr_t)l, 0, 0);
#else
  *(uint4*)l = *(const uint4*)g;
#endif
}

__device__ __forceinline__ void async_wait0() {
#if HAVE_ASYNC_LDS
#  if __has_builtin(__builtin_amdgcn_s_wait_asynccnt)
  __builtin_amdgcn_s_wait_asynccnt(0);
#  else
  asm volatile("s_wait_asynccnt 0x0" ::: "memory");
#  endif
#endif
}

__device__ __forceinline__ __bf16 f2bf(float f) {
  unsigned u = __float_as_uint(f);
  unsigned r = (u + 0x7FFFu + ((u >> 16) & 1u)) >> 16;
  unsigned short us = (unsigned short)r;
  __bf16 b;
  __builtin_memcpy(&b, &us, 2);
  return b;
}

// Build a 16-element bf16 fragment from two 16-byte chunks (LDS or global).
__device__ __forceinline__ v16bf load_frag(const __bf16* p0, const __bf16* p1) {
  v16bf r;
  ((uint4*)&r)[0] = *(const uint4*)p0;
  ((uint4*)&r)[1] = *(const uint4*)p1;
  return r;
}

// ---------------- weight transpose: W[K][N] fp32 -> Wt[N][K] bf16 ----------------
__global__ void k_transpose_bf16(const float* __restrict__ W, __bf16* __restrict__ Wt,
                                 int K, int N) {
  __shared__ float tile[32][33];
  int n0 = blockIdx.x * 32;
  int k0 = blockIdx.y * 32;
  int tx = threadIdx.x & 31, ty = threadIdx.x >> 5;   // 256 thr: ty 0..7
  #pragma unroll
  for (int i = ty; i < 32; i += 8)
    tile[i][tx] = W[(size_t)(k0 + i) * N + n0 + tx];
  __syncthreads();
  #pragma unroll
  for (int i = ty; i < 32; i += 8)
    Wt[(size_t)(n0 + i) * K + k0 + tx] = f2bf(tile[tx][i]);
}

// ---------------- embedding gather -> bf16 activations ----------------
__global__ void k_embed(const int* __restrict__ inputs, const float* __restrict__ emb,
                        __bf16* __restrict__ xb) {
  int m = blockIdx.x;                 // 0..2047  (= b*S + s)
  int tok = inputs[m];
  const float* src = emb + (size_t)tok * E_;
  __bf16* dst = xb + (size_t)m * E_;
  for (int e = threadIdx.x; e < E_; e += blockDim.x) dst[e] = f2bf(src[e]);
}

// ---------------- LSTM state init ----------------
__global__ void k_init(const float* __restrict__ h0, const float* __restrict__ c0,
                       float* __restrict__ hf, __bf16* __restrict__ hb,
                       float* __restrict__ c) {
  int i = blockIdx.x * blockDim.x + threadIdx.x;
  if (i < B_ * U_) {
    float h = h0[i];
    hf[i] = h;
    hb[i] = f2bf(h);
    c[i]  = c0[i];
  }
}

// ---------------- generic WMMA bf16 GEMM: C = A[M][K] * Bt[N][K]^T + bias ----------------
// Block tile 128x128, K step 32, double-buffered async LDS staging.
// 8 waves: 4(M) x 2(N), each wave 32x64 = 2x4 wmma tiles.
template<bool RELU, bool OUTBF>
__global__ void k_gemm(const __bf16* __restrict__ A, const __bf16* __restrict__ Bt,
                       const float* __restrict__ bias, void* __restrict__ Cout,
                       int Ndim, int Kdim) {
  __shared__ __align__(16) __bf16 lA[2][128][32];
  __shared__ __align__(16) __bf16 lB[2][128][32];
  const int n0 = blockIdx.x * 128;
  const int m0 = blockIdx.y * 128;
  const int tid = threadIdx.x;
  const int wave = tid >> 5, lane = tid & 31;
  const int wm = (wave & 3) * 32;     // wave M offset
  const int wn = (wave >> 2) * 64;    // wave N offset
  const int lhalf = lane >> 4;        // lane half (K split per ISA layout)
  const int lmod  = lane & 15;
  const int lrow = tid >> 1;          // staging: 2 threads per row
  const int lk   = (tid & 1) * 16;    // 16 bf16 = 32B per thread
  v8f acc[2][4] = {};

  auto stage = [&](int buf, int kk) {
    copy16_to_lds(A  + (size_t)(m0 + lrow) * Kdim + kk + lk, &lA[buf][lrow][lk]);
    copy16_to_lds(Bt + (size_t)(n0 + lrow) * Kdim + kk + lk, &lB[buf][lrow][lk]);
  };

  stage(0, 0);
  int cur = 0;
  for (int kk = 0; kk < Kdim; kk += 32) {
    async_wait0();        // current tile's async copies have landed in LDS
    __syncthreads();      // visible to all waves; prev compute on other buffer done
    if (kk + 32 < Kdim) stage(cur ^ 1, kk + 32);   // overlap next tile with compute

    v16bf afr[2], bfr[4];
    #pragma unroll
    for (int i = 0; i < 2; ++i) {
      // A 16x32 bf16 layout: lane<16 row=lane, K=e (+8 for lanes>=16); chunk1 at K+16
      const __bf16* p = &lA[cur][wm + i * 16 + lmod][lhalf * 8];
      afr[i] = load_frag(p, p + 16);
    }
    #pragma unroll
    for (int j = 0; j < 4; ++j) {
      // B 32x16 bf16 layout: N = lane%16, K = e (+8 for lanes>=16); chunk1 at K+16
      const __bf16* p = &lB[cur][wn + j * 16 + lmod][lhalf * 8];
      bfr[j] = load_frag(p, p + 16);
    }
    #pragma unroll
    for (int i = 0; i < 2; ++i)
      #pragma unroll
      for (int j = 0; j < 4; ++j)
        acc[i][j] = __builtin_amdgcn_wmma_f32_16x16x32_bf16(
            false, afr[i], false, bfr[j], (short)0, acc[i][j], false, false);
    cur ^= 1;
  }

  // C/D f32 layout: element r -> M = r + lanehalf*8, N = lane%16
  #pragma unroll
  for (int i = 0; i < 2; ++i) {
    int mbase = m0 + wm + i * 16 + lhalf * 8;
    #pragma unroll
    for (int j = 0; j < 4; ++j) {
      int col = n0 + wn + j * 16 + lmod;
      float bv = bias[col];
      #pragma unroll
      for (int r = 0; r < 8; ++r) {
        float v = acc[i][j][r] + bv;
        if (RELU) v = fmaxf(v, 0.f);
        size_t idx = (size_t)(mbase + r) * Ndim + col;
        if (OUTBF) ((__bf16*)Cout)[idx] = f2bf(v);
        else       ((float*)Cout)[idx]  = v;
      }
    }
  }
}

// ---------------- fused LSTM step: z = xg[:,s,:] + h@Wh ; gates ; mask ; BN-out ----------------
// Grid: 16 blocks, block handles u-tile of 64. 8 waves: wave -> (gate g = w>>1, half = w&1).
__global__ void k_lstm_step(const __bf16* __restrict__ hprev_bf,
                            const float*  __restrict__ hprev_f,
                            float* __restrict__ cbuf,
                            const __bf16* __restrict__ Wht,   // [4096][1024] bf16
                            const float*  __restrict__ xg,    // [2048][4096] fp32 (bias included)
                            const int*    __restrict__ inputs,
                            const float* __restrict__ gamma, const float* __restrict__ beta,
                            const float* __restrict__ mmean, const float* __restrict__ mvar,
                            float* __restrict__ hnext_f, __bf16* __restrict__ hnext_bf,
                            __bf16* __restrict__ hbn,
                            int s) {
  __shared__ __align__(16) __bf16 hA[B_][U_];   // 32 KB: whole h in bf16
  __shared__ float zl[4][B_][64];               // 16 KB: z tiles for gate combine
  const int tid = threadIdx.x;

  // stage h_prev (bf16) into LDS via async DMA: 16384 bf16 = 2048 x 16B
  {
    const __bf16* src = hprev_bf;
    __bf16* dst = &hA[0][0];
    for (int i = tid; i < (B_ * U_) / 8; i += 256)
      copy16_to_lds(src + i * 8, dst + i * 8);
  }
  async_wait0();
  __syncthreads();

  const int wave = tid >> 5, lane = tid & 31;
  const int g = wave >> 1, half = wave & 1;
  const int u0 = blockIdx.x * 64;
  const int lhalf = lane >> 4, lmod = lane & 15;
  v8f acc[2] = {};

  for (int kk = 0; kk < U_; kk += 32) {
    const __bf16* pa = &hA[lmod][kk + lhalf * 8];
    v16bf afr = load_frag(pa, pa + 16);
    #pragma unroll
    for (int t = 0; t < 2; ++t) {
      int col = g * U_ + u0 + half * 32 + t * 16 + lmod;
      const __bf16* pb = Wht + (size_t)col * U_ + kk + lhalf * 8;  // L2-resident
      v16bf bfr = load_frag(pb, pb + 16);
      acc[t] = __builtin_amdgcn_wmma_f32_16x16x32_bf16(
          false, afr, false, bfr, (short)0, acc[t], false, false);
    }
  }

  // z = acc + xg ; stash to LDS for cross-gate combine
  #pragma unroll
  for (int t = 0; t < 2; ++t) {
    int ul  = half * 32 + t * 16 + lmod;
    int col = g * U_ + u0 + ul;
    #pragma unroll
    for (int r = 0; r < 8; ++r) {
      int b = r + lhalf * 8;
      zl[g][b][ul] = acc[t][r] + xg[(size_t)(b * S_ + s) * G4 + col];
    }
  }
  __syncthreads();

  // gate combine + mask + BN; 16*64 = 1024 (b,u) pairs
  for (int p = tid; p < B_ * 64; p += 256) {
    int b = p >> 6, ul = p & 63, u = u0 + ul;
    float zi = zl[0][b][ul], zf = zl[1][b][ul], zc = zl[2][b][ul], zo = zl[3][b][ul];
    float ig = 1.f / (1.f + __expf(-zi));
    float fg = 1.f / (1.f + __expf(-zf));
    float og = 1.f / (1.f + __expf(-zo));
    float cc = fmaxf(zc, 0.f);
    int idx = b * U_ + u;
    float cold = cbuf[idx];
    float cnew = fg * cold + ig * cc;
    float hnew = og * fmaxf(cnew, 0.f);
    if (inputs[b * S_ + s] == 0) { hnew = hprev_f[idx]; cnew = cold; }
    cbuf[idx]    = cnew;
    hnext_f[idx] = hnew;
    hnext_bf[idx] = f2bf(hnew);
    float hb = (hnew - mmean[u]) * __frsqrt_rn(mvar[u] + 1e-3f) * gamma[u] + beta[u];
    hbn[(size_t)(b * S_ + s) * U_ + u] = f2bf(hb);
  }
}

// ---------------- softmax over V=32000, in place on d_out ----------------
__global__ void k_softmax(float* __restrict__ out) {
  __shared__ float red[256];
  int row = blockIdx.x;
  float* p = out + (size_t)row * V_;
  int tid = threadIdx.x;
  float mx = -3.4e38f;
  for (int i = tid; i < V_; i += 256) mx = fmaxf(mx, p[i]);
  red[tid] = mx; __syncthreads();
  for (int st = 128; st > 0; st >>= 1) {
    if (tid < st) red[tid] = fmaxf(red[tid], red[tid + st]);
    __syncthreads();
  }
  mx = red[0]; __syncthreads();
  float sum = 0.f;
  for (int i = tid; i < V_; i += 256) {
    float e = __expf(p[i] - mx);
    p[i] = e;
    sum += e;
  }
  red[tid] = sum; __syncthreads();
  for (int st = 128; st > 0; st >>= 1) {
    if (tid < st) red[tid] += red[tid + st];
    __syncthreads();
  }
  float inv = 1.f / red[0];
  for (int i = tid; i < V_; i += 256) p[i] *= inv;
}

extern "C" void kernel_launch(void* const* d_in, const int* in_sizes, int n_in,
                              void* d_out, int out_size, void* d_ws, size_t ws_size,
                              hipStream_t stream) {
  const int*   inputs = (const int*)  d_in[0];
  const float* enc_h  = (const float*)d_in[1];
  const float* enc_c  = (const float*)d_in[2];
  const float* emb    = (const float*)d_in[3];
  const float* Wx     = (const float*)d_in[4];
  const float* Wh     = (const float*)d_in[5];
  const float* bb     = (const float*)d_in[6];
  const float* gamma  = (const float*)d_in[7];
  const float* beta   = (const float*)d_in[8];
  const float* mmean  = (const float*)d_in[9];
  const float* mvar   = (const float*)d_in[10];
  const float* W1     = (const float*)d_in[11];
  const float* b1     = (const float*)d_in[12];
  const float* W2     = (const float*)d_in[13];
  const float* b2     = (const float*)d_in[14];

  char* ws = (char*)d_ws;
  size_t off = 0;
  auto alloc = [&](size_t bytes) -> void* {
    void* p = ws + off;
    off = (off + bytes + 255) & ~(size_t)255;
    return p;
  };
  __bf16* Wxt  = (__bf16*)alloc((size_t)G4 * E_ * 2);   //  4 MB
  __bf16* Wht  = (__bf16*)alloc((size_t)G4 * U_ * 2);   //  8 MB (L2-resident across steps)
  __bf16* W1t  = (__bf16*)alloc((size_t)U_ * U_ * 2);   //  2 MB
  __bf16* W2t  = (__bf16*)alloc((size_t)V_ * U_ * 2);   // 64 MB (fits 192 MB L2)
  __bf16* xb   = (__bf16*)alloc((size_t)M_ * E_ * 2);   //  2 MB
  float*  xg   = (float*) alloc((size_t)M_ * G4 * 4);   // 32 MB
  __bf16* hbn  = (__bf16*)alloc((size_t)M_ * U_ * 2);   //  4 MB
  __bf16* d1   = (__bf16*)alloc((size_t)M_ * U_ * 2);   //  4 MB
  float*  hf0  = (float*) alloc((size_t)B_ * U_ * 4);
  float*  hf1  = (float*) alloc((size_t)B_ * U_ * 4);
  __bf16* hb0  = (__bf16*)alloc((size_t)B_ * U_ * 2);
  __bf16* hb1  = (__bf16*)alloc((size_t)B_ * U_ * 2);
  float*  cbuf = (float*) alloc((size_t)B_ * U_ * 4);
  float* hfp[2] = {hf0, hf1};
  __bf16* hbp[2] = {hb0, hb1};

  // 1) weight transposes (fp32 -> bf16 [N][K])
  k_transpose_bf16<<<dim3(G4 / 32, E_ / 32), 256, 0, stream>>>(Wx, Wxt, E_, G4);
  k_transpose_bf16<<<dim3(G4 / 32, U_ / 32), 256, 0, stream>>>(Wh, Wht, U_, G4);
  k_transpose_bf16<<<dim3(U_ / 32, U_ / 32), 256, 0, stream>>>(W1, W1t, U_, U_);
  k_transpose_bf16<<<dim3(V_ / 32, U_ / 32), 256, 0, stream>>>(W2, W2t, U_, V_);

  // 2) embedding gather
  k_embed<<<M_, 256, 0, stream>>>(inputs, emb, xb);

  // 3) xg = x @ Wx + b   (2048 x 4096, K=512)
  k_gemm<false, false><<<dim3(G4 / 128, M_ / 128), 256, 0, stream>>>(
      xb, Wxt, bb, xg, G4, E_);

  // 4) LSTM scan (128 sequential steps; Wh bf16 stays in L2)
  k_init<<<(B_ * U_ + 255) / 256, 256, 0, stream>>>(enc_h, enc_c, hf0, hb0, cbuf);
  for (int s = 0; s < S_; ++s) {
    int cur = s & 1, nxt = cur ^ 1;
    k_lstm_step<<<U_ / 64, 256, 0, stream>>>(
        hbp[cur], hfp[cur], cbuf, Wht, xg, inputs,
        gamma, beta, mmean, mvar,
        hfp[nxt], hbp[nxt], hbn, s);
  }

  // 5) d1 = relu(hbn @ W1 + b1) -> bf16
  k_gemm<true, true><<<dim3(U_ / 128, M_ / 128), 256, 0, stream>>>(
      hbn, W1t, b1, d1, U_, U_);

  // 6) logits = d1 @ W2 + b2 -> fp32 in d_out  (dominant GEMM: 2048x32000, K=1024)
  k_gemm<false, false><<<dim3(V_ / 128, M_ / 128), 256, 0, stream>>>(
      d1, W2t, b2, (float*)d_out, V_, U_);

  // 7) softmax rows in place
  k_softmax<<<M_, 256, 0, stream>>>((float*)d_out);
}
